// ResAttention_2482491097650
// MI455X (gfx1250) — compile-verified, matrix operand order checked by reading, not run on hardware
//
#include <hip/hip_runtime.h>
#include <cstdint>
#include <cstddef>

// Problem constants (match reference)
#define B_    8
#define S_    2048
#define NIN_  256
#define NOUT_ 512
#define M_    (B_ * S_)     // 16384 token rows
#define EPS_  1e-5f

typedef __attribute__((ext_vector_type(16))) __bf16 v16bf;
typedef __attribute__((ext_vector_type(8)))  __bf16 v8bf;
typedef __attribute__((ext_vector_type(8)))  float  v8f;

static __device__ __forceinline__ v16bf cat8(v8bf lo, v8bf hi) {
    return __builtin_shufflevector(lo, hi, 0,1,2,3,4,5,6,7,8,9,10,11,12,13,14,15);
}
// Load a 16-element bf16 fragment for the CDNA5 16-bit WMMA layout:
// lane half h owns K = {h*8..h*8+7} and {16+h*8..16+h*8+7}  -> two b128 loads.
static __device__ __forceinline__ v16bf load_frag(const __bf16* p) {
    v8bf lo = *(const v8bf*)(p);
    v8bf hi = *(const v8bf*)(p + 16);
    return cat8(lo, hi);
}
// LDS byte offset of a shared-memory pointer (ISA: LDS aperture maps addr[31:0]
// to the wave's LDS allocation offset).
static __device__ __forceinline__ unsigned lds_off(const void* p) {
    return (unsigned)(uintptr_t)p;
}

// ---------------------------------------------------------------- f32 -> bf16
__global__ __launch_bounds__(256)
void cvt_bf16_kernel(const float* __restrict__ in, __bf16* __restrict__ out, size_t n) {
    size_t i = ((size_t)blockIdx.x * blockDim.x + threadIdx.x) * 4;
    if (i + 3 < n) {
        float4 f = *(const float4*)(in + i);
        out[i + 0] = (__bf16)f.x;
        out[i + 1] = (__bf16)f.y;
        out[i + 2] = (__bf16)f.z;
        out[i + 3] = (__bf16)f.w;
    } else {
        for (; i < n; ++i) out[i] = (__bf16)in[i];
    }
}

// ------------------------------------------------- generic bf16 WMMA GEMM
// C[M,N] = A[M,K] @ W[N,K]^T + bias.  Block = 128 rows x 64 cols, 8 waves.
// The 64xK weight panel (contiguous in W) is staged into LDS once per block
// via CDNA5 async-to-LDS copies; B-fragments then come from LDS (ds_load_b128),
// A-fragments stream from global (each wave owns disjoint rows).
__global__ __launch_bounds__(256)
void gemm_bf16_wmma(const __bf16* __restrict__ A, const __bf16* __restrict__ W,
                    const float* __restrict__ bias,
                    float* __restrict__ outF, __bf16* __restrict__ outB,
                    const float* __restrict__ resid,
                    int M, int N, int K) {
    __shared__ __attribute__((aligned(16))) __bf16 sW[64 * 512];   // 64 KB max panel

    const int wave = threadIdx.x >> 5;
    const int lane = threadIdx.x & 31;
    const int row  = lane & 15;   // A row / B column within 16-wide tile
    const int hh   = lane >> 4;   // K-half select
    const int m0   = (blockIdx.y * 8 + wave) * 16;
    const int n0   = blockIdx.x * 64;

    // ---- async-stage the weight panel: one contiguous 64*K*2-byte region ----
    {
        const char* src = (const char*)(W + (size_t)n0 * K);
        const unsigned bytes = 64u * (unsigned)K * 2u;
        const unsigned base  = lds_off(&sW[0]);
        for (unsigned off = threadIdx.x * 16u; off < bytes; off += 4096u) {
            asm volatile("global_load_async_to_lds_b128 %0, %1, %2"
                         :: "v"(base + off), "v"(off), "s"(src)
                         : "memory");
        }
        asm volatile("s_wait_asynccnt 0" ::: "memory");
    }
    __syncthreads();
    if (m0 >= M) return;

    v8f acc0 = {}, acc1 = {}, acc2 = {}, acc3 = {};

    const __bf16* ap  = A + (size_t)(m0 + row) * K + hh * 8;
    const __bf16* wp0 = sW + (size_t)( 0 + row) * K + hh * 8;
    const __bf16* wp1 = sW + (size_t)(16 + row) * K + hh * 8;
    const __bf16* wp2 = sW + (size_t)(32 + row) * K + hh * 8;
    const __bf16* wp3 = sW + (size_t)(48 + row) * K + hh * 8;

    for (int kb = 0; kb < K; kb += 32) {
        v16bf af = load_frag(ap + kb);
        v16bf b0 = load_frag(wp0 + kb);   // ds_load_b128 x2
        v16bf b1 = load_frag(wp1 + kb);
        v16bf b2 = load_frag(wp2 + kb);
        v16bf b3 = load_frag(wp3 + kb);
        acc0 = __builtin_amdgcn_wmma_f32_16x16x32_bf16(false, af, false, b0, (short)0, acc0, false, false);
        acc1 = __builtin_amdgcn_wmma_f32_16x16x32_bf16(false, af, false, b1, (short)0, acc1, false, false);
        acc2 = __builtin_amdgcn_wmma_f32_16x16x32_bf16(false, af, false, b2, (short)0, acc2, false, false);
        acc3 = __builtin_amdgcn_wmma_f32_16x16x32_bf16(false, af, false, b3, (short)0, acc3, false, false);
    }

    // D layout: lane n = lane&15, rows m = hh*8 + v  (v = VGPR index 0..7)
    v8f accs[4] = {acc0, acc1, acc2, acc3};
#pragma unroll
    for (int t = 0; t < 4; ++t) {
        const int n  = n0 + t * 16 + row;
        const float bv = bias ? bias[n] : 0.0f;
#pragma unroll
        for (int v = 0; v < 8; ++v) {
            const size_t idx = (size_t)(m0 + hh * 8 + v) * N + n;
            float val = accs[t][v] + bv;
            if (resid) val += resid[idx];
            if (outF) outF[idx] = val;
            if (outB) outB[idx] = (__bf16)val;
        }
    }
}

// ------------------------------- transpose x[B,S,D] f32 -> xT[B,D,S] bf16
__global__ __launch_bounds__(256)
void transpose_bf16(const float* __restrict__ x, __bf16* __restrict__ xT) {
    __shared__ __bf16 tile[32][33];
    const int b  = blockIdx.z;
    const int s0 = blockIdx.x * 32;
    const int d0 = blockIdx.y * 32;
    const int tx = threadIdx.x & 31;
    const int ty = threadIdx.x >> 5;
    for (int i = ty; i < 32; i += 8)
        tile[i][tx] = (__bf16)x[((size_t)(b * S_ + s0 + i)) * NOUT_ + d0 + tx];
    __syncthreads();
    for (int i = ty; i < 32; i += 8)
        xT[((size_t)(b * NOUT_ + d0 + i)) * S_ + s0 + tx] = tile[tx][i];
}

// --------------------------------------------------- fused masked attention
// One block per (batch, 16-query tile).  Scores for all S keys kept in a
// 64KB LDS tile (bf16), softmax in place, then h = P @ x with each wave
// owning a 64-wide slice of d.
__global__ __launch_bounds__(256)
void attn_kernel(const __bf16* __restrict__ q, const __bf16* __restrict__ k,
                 const __bf16* __restrict__ xT, const int* __restrict__ lengths,
                 float* __restrict__ h) {
    __shared__ __attribute__((aligned(16))) __bf16 sc[16 * S_];   // 64 KB

    const int b    = blockIdx.y;
    const int q0   = blockIdx.x * 16;
    const int wave = threadIdx.x >> 5;
    const int lane = threadIdx.x & 31;
    const int row  = lane & 15;
    const int hh   = lane >> 4;
    const int len  = lengths[b];
    const float scale = 0.04419417382415922f;   // 1/sqrt(512)
    const size_t tokbase = (size_t)b * S_ * NOUT_;

    // Cache this block's 16-query tile in registers: 16 K-step fragments.
    v16bf qf[16];
    const __bf16* qp = q + tokbase + (size_t)(q0 + row) * NOUT_ + hh * 8;
#pragma unroll
    for (int ks = 0; ks < 16; ++ks) qf[ks] = load_frag(qp + ks * 32);

    // Pass 1: scores = q @ k^T  (wave w handles key tiles w, w+8, ...)
    for (int nt = wave; nt < S_ / 16; nt += 8) {
        const int kk0 = nt * 16;
        v8f acc = {};
        const __bf16* kp = k + tokbase + (size_t)(kk0 + row) * NOUT_ + hh * 8;
        // prefetch the row this lane touches 8 tiles from now (global_prefetch_b8)
        __builtin_prefetch(kp + (size_t)8 * 16 * NOUT_, 0, 0);
#pragma unroll
        for (int ks = 0; ks < 16; ++ks) {
            v16bf kf = load_frag(kp + ks * 32);
            acc = __builtin_amdgcn_wmma_f32_16x16x32_bf16(false, qf[ks], false, kf,
                                                          (short)0, acc, false, false);
        }
        const int key = kk0 + row;
        const float mval = (key < len) ? scale : 0.0f;
        const float madd = (key < len) ? 0.0f : -__builtin_inff();
#pragma unroll
        for (int v = 0; v < 8; ++v)
            sc[(hh * 8 + v) * S_ + key] = (__bf16)(acc[v] * mval + madd);
    }
    __syncthreads();

    // Softmax in LDS: 16 lanes per query row.
    {
        const int r   = threadIdx.x >> 4;
        const int sub = threadIdx.x & 15;
        __bf16* rp = sc + r * S_;
        float mx = -__builtin_inff();
        for (int j = sub; j < S_; j += 16) mx = fmaxf(mx, (float)rp[j]);
#pragma unroll
        for (int o = 8; o >= 1; o >>= 1) mx = fmaxf(mx, __shfl_xor(mx, o, 16));
        float sum = 0.0f;
        for (int j = sub; j < S_; j += 16) sum += __expf((float)rp[j] - mx);
#pragma unroll
        for (int o = 8; o >= 1; o >>= 1) sum += __shfl_xor(sum, o, 16);
        const float inv = 1.0f / sum;
        for (int j = sub; j < S_; j += 16)
            rp[j] = (__bf16)(__expf((float)rp[j] - mx) * inv);
    }
    __syncthreads();

    // Pass 2: h = P @ x ; wave owns d-slice [wave*64, wave*64+64).
    const int d0 = wave * 64;
    v8f hacc0 = {}, hacc1 = {}, hacc2 = {}, hacc3 = {};
    const __bf16* xp0 = xT + (size_t)b * NOUT_ * S_ + (size_t)(d0 +  0 + row) * S_ + hh * 8;
    const __bf16* xp1 = xp0 + (size_t)16 * S_;
    const __bf16* xp2 = xp0 + (size_t)32 * S_;
    const __bf16* xp3 = xp0 + (size_t)48 * S_;
    const __bf16* pp  = sc + row * S_ + hh * 8;
    for (int kb = 0; kb < S_; kb += 32) {
        v16bf pf = load_frag(pp + kb);       // ds_load_b128 x2 from LDS
        hacc0 = __builtin_amdgcn_wmma_f32_16x16x32_bf16(false, pf, false, load_frag(xp0 + kb), (short)0, hacc0, false, false);
        hacc1 = __builtin_amdgcn_wmma_f32_16x16x32_bf16(false, pf, false, load_frag(xp1 + kb), (short)0, hacc1, false, false);
        hacc2 = __builtin_amdgcn_wmma_f32_16x16x32_bf16(false, pf, false, load_frag(xp2 + kb), (short)0, hacc2, false, false);
        hacc3 = __builtin_amdgcn_wmma_f32_16x16x32_bf16(false, pf, false, load_frag(xp3 + kb), (short)0, hacc3, false, false);
    }
    float* hp = h + ((size_t)b * S_ + q0) * NOUT_;
    v8f haccs[4] = {hacc0, hacc1, hacc2, hacc3};
#pragma unroll
    for (int t = 0; t < 4; ++t) {
        const int n = d0 + t * 16 + row;
#pragma unroll
        for (int v = 0; v < 8; ++v)
            hp[(size_t)(hh * 8 + v) * NOUT_ + n] = haccs[t][v];
    }
}

// --------------------------------------------- LayerNorm + ReLU -> bf16
__global__ __launch_bounds__(256)
void ln_relu_bf16(const float* __restrict__ h, const float* __restrict__ g,
                  const float* __restrict__ beta, __bf16* __restrict__ out, int M) {
    const int wave = threadIdx.x >> 5;
    const int lane = threadIdx.x & 31;
    const int rowid = blockIdx.x * 8 + wave;
    if (rowid >= M) return;
    const float* hp = h + (size_t)rowid * NOUT_;

    float vals[16];
    float s = 0.0f;
#pragma unroll
    for (int i = 0; i < 16; ++i) { vals[i] = hp[lane + 32 * i]; s += vals[i]; }
#pragma unroll
    for (int o = 16; o >= 1; o >>= 1) s += __shfl_xor(s, o, 32);
    const float mu = s * (1.0f / NOUT_);
    float var = 0.0f;
#pragma unroll
    for (int i = 0; i < 16; ++i) { const float d = vals[i] - mu; var += d * d; }
#pragma unroll
    for (int o = 16; o >= 1; o >>= 1) var += __shfl_xor(var, o, 32);
    const float inv = rsqrtf(var * (1.0f / NOUT_) + EPS_);

    __bf16* op = out + (size_t)rowid * NOUT_;
#pragma unroll
    for (int i = 0; i < 16; ++i) {
        const int c = lane + 32 * i;
        const float y = (vals[i] - mu) * inv * g[c] + beta[c];
        op[c] = (__bf16)fmaxf(y, 0.0f);
    }
}

// =============================================================== launcher
extern "C" void kernel_launch(void* const* d_in, const int* in_sizes, int n_in,
                              void* d_out, int out_size, void* d_ws, size_t ws_size,
                              hipStream_t stream) {
    (void)in_sizes; (void)n_in; (void)out_size; (void)ws_size;
    const float* seq     = (const float*)d_in[0];
    const int*   lengths = (const int*)  d_in[1];
    const float* W0      = (const float*)d_in[2];
    const float* b0      = (const float*)d_in[3];
    const float* Wq      = (const float*)d_in[4];
    const float* bq      = (const float*)d_in[5];
    const float* Wk      = (const float*)d_in[6];
    const float* bk      = (const float*)d_in[7];
    const float* ln_g    = (const float*)d_in[8];
    const float* ln_b    = (const float*)d_in[9];
    const float* Wl      = (const float*)d_in[10];
    const float* bl      = (const float*)d_in[11];
    float* out = (float*)d_out;

    char* ws = (char*)d_ws;
    size_t off = 0;
    auto take = [&](size_t bytes) -> char* {
        char* p = ws + off;
        off += (bytes + 255) & ~(size_t)255;
        return p;
    };
    __bf16* seqb = (__bf16*)take((size_t)M_ * NIN_ * 2);
    __bf16* W0b  = (__bf16*)take((size_t)NOUT_ * NIN_ * 2);
    __bf16* Wqb  = (__bf16*)take((size_t)NOUT_ * NOUT_ * 2);
    __bf16* Wkb  = (__bf16*)take((size_t)NOUT_ * NOUT_ * 2);
    __bf16* Wlb  = (__bf16*)take((size_t)2 * NOUT_ * NOUT_ * 2);
    float*  x    = (float*) take((size_t)M_ * NOUT_ * 4);
    __bf16* xb   = (__bf16*)take((size_t)M_ * NOUT_ * 2);
    __bf16* xT   = (__bf16*)take((size_t)M_ * NOUT_ * 2);
    __bf16* qb   = (__bf16*)take((size_t)M_ * NOUT_ * 2);
    __bf16* kb   = (__bf16*)take((size_t)M_ * NOUT_ * 2);
    float*  h1   = (float*) take((size_t)M_ * NOUT_ * 4);
    __bf16* ab   = (__bf16*)take((size_t)M_ * NOUT_ * 2);
    float*  h2   = (float*) take((size_t)M_ * NOUT_ * 4);

    auto cvt = [&](const float* src, __bf16* dst, size_t n) {
        const unsigned blocks = (unsigned)((n / 4 + 255) / 256);
        cvt_bf16_kernel<<<dim3(blocks), dim3(256), 0, stream>>>(src, dst, n);
    };
    cvt(seq, seqb, (size_t)M_ * NIN_);
    cvt(W0,  W0b,  (size_t)NOUT_ * NIN_);
    cvt(Wq,  Wqb,  (size_t)NOUT_ * NOUT_);
    cvt(Wk,  Wkb,  (size_t)NOUT_ * NOUT_);
    cvt(Wl,  Wlb,  (size_t)2 * NOUT_ * NOUT_);

    const dim3 blk(256);
    const dim3 ggrid(NOUT_ / 64, M_ / 128);   // (8, 128)

    // x = seq @ W0^T + b0   (also bf16 copy for downstream matmuls)
    gemm_bf16_wmma<<<ggrid, blk, 0, stream>>>(seqb, W0b, b0, x, xb, nullptr, M_, NOUT_, NIN_);
    // xT (bf16, [B, D, S]) for the P@x stage
    transpose_bf16<<<dim3(S_ / 32, NOUT_ / 32, B_), blk, 0, stream>>>(x, xT);
    // q / k projections
    gemm_bf16_wmma<<<ggrid, blk, 0, stream>>>(xb, Wqb, bq, nullptr, qb, nullptr, M_, NOUT_, NOUT_);
    gemm_bf16_wmma<<<ggrid, blk, 0, stream>>>(xb, Wkb, bk, nullptr, kb, nullptr, M_, NOUT_, NOUT_);
    // fused masked attention -> h1
    attn_kernel<<<dim3(S_ / 16, B_), blk, 0, stream>>>(qb, kb, xT, lengths, h1);
    // residual stack layer 0
    ln_relu_bf16<<<dim3(M_ / 8), blk, 0, stream>>>(h1, ln_g, ln_b, ab, M_);
    gemm_bf16_wmma<<<ggrid, blk, 0, stream>>>(ab, Wlb, bl, h2, nullptr, nullptr, M_, NOUT_, NOUT_);
    // residual stack layer 1 (+ final residual x), writes d_out
    ln_relu_bf16<<<dim3(M_ / 8), blk, 0, stream>>>(h2, ln_g + NOUT_, ln_b + NOUT_, ab, M_);
    gemm_bf16_wmma<<<ggrid, blk, 0, stream>>>(ab, Wlb + (size_t)NOUT_ * NOUT_, bl + NOUT_,
                                              out, nullptr, x, M_, NOUT_, NOUT_);
}
